// GCN_19791209300146
// MI455X (gfx1250) — matrix operand order repaired and checked by reading, not compiled, not run on hardware
//
#include <hip/hip_runtime.h>
#include <math.h>

#define Nn   20000
#define Mm   2048
#define Ee   320000
#define Ff   512
#define HIDd 256
#define OUTc 2

typedef __attribute__((ext_vector_type(16))) _Float16 v16h;
typedef __attribute__((ext_vector_type(8)))  float    v8f;

// ---------- ordered-uint encoding for float atomicMax ----------
__device__ __forceinline__ unsigned fenc(float f) {
  unsigned u = __float_as_uint(f);
  return (u & 0x80000000u) ? ~u : (u | 0x80000000u);
}
__device__ __forceinline__ float fdec(unsigned u) {
  u = (u & 0x80000000u) ? (u & 0x7FFFFFFFu) : ~u;
  return __uint_as_float(u);
}
#define ENC_NEG_INF 0x007FFFFFu  // fenc(-inf), host-side constant

// ---------- weight pre-swizzle: W[K x Nc] f32 -> f16 WMMA B-fragments ----------
// Layout: Bswz[(ct*(K/32) + kc)*32 + lane] is the v16h fragment for column tile
// ct, k-chunk kc, lane. B-fragment element idx (0..15): n = ct*16 + (lane&15),
// k = kc*32 + (lane>>4)*16 + idx  (contiguous k order, same as the A tile).
__global__ void prep_b_kernel(const float* __restrict__ W, _Float16* __restrict__ Bswz,
                              int K, int Nc) {
  int t = blockIdx.x * blockDim.x + threadIdx.x;
  int total = (Nc / 16) * (K / 32) * 32;
  if (t >= total) return;
  int lane = t & 31;
  int rest = t >> 5;
  int kc = rest % (K / 32);
  int ct = rest / (K / 32);
  int n  = ct * 16 + (lane & 15);
  int hs = lane >> 4;
  v16h vv;
#pragma unroll
  for (int idx = 0; idx < 16; ++idx) {
    int k = kc * 32 + hs * 16 + idx;
    vv[idx] = (_Float16)W[(size_t)k * Nc + n];
  }
  *((v16h*)(Bswz + (size_t)t * 16)) = vv;
}

// ---------- WMMA GEMM: C[Mr x Nc] = act(A[Mr x 512] @ B + bias) ----------
// K fixed at 512. block = 256 threads = 8 waves, covers 16 rows x all Nc.
// grid = (Mr/16).  Mr, Nc must be multiples of 16 (true here: 20000/2048, 512/256).
// A tile staged in LDS pre-swizzled to A fragments, contiguous k order as in Bswz:
//   lane = (m&15) + 16*(kk>>4),  idx = kk&15,  kk = k%32.
template <int ACT, int ADD>
__global__ __launch_bounds__(256) void gemm_wmma_kernel(
    const float* __restrict__ A, const _Float16* __restrict__ Bswz,
    const float* __restrict__ bias, float* __restrict__ C,
    int Nc, float slope) {
  __shared__ _Float16 __attribute__((aligned(32))) As[16 * 32 * 16];  // 16KB

  const int rowBase = blockIdx.x * 16;

  // cooperative stage + swizzle of the 16 x 512 A tile (k-major for coalescing)
  for (int i = threadIdx.x; i < 16 * 512; i += 256) {
    int m = i >> 9;         // 0..15
    int k = i & 511;        // 0..511
    float v = A[(size_t)(rowBase + m) * 512 + k];
    int kc   = k >> 5;
    int kk   = k & 31;
    int hs   = kk >> 4;
    int lane = m + (hs << 4);
    int idx  = kk & 15;
    As[(((kc << 5) + lane) << 4) + idx] = (_Float16)v;
  }
  __syncthreads();

  const int lane = threadIdx.x & 31;
  const int wave = threadIdx.x >> 5;
  const int hs   = lane >> 4;
  const int nTiles = Nc >> 4;

  for (int ct = wave; ct < nTiles; ct += 8) {
    v8f acc = {};
#pragma unroll 4
    for (int kc = 0; kc < 16; ++kc) {
      v16h af = *((const v16h*)&As[(((kc << 5) + lane) << 4)]);
      v16h bf = *((const v16h*)(Bswz + ((size_t)(ct * 16 + kc) * 32 + lane) * 16));
      acc = __builtin_amdgcn_wmma_f32_16x16x32_f16(false, af, false, bf,
                                                   (short)0, acc, false, false);
    }
    const int ncol = ct * 16 + (lane & 15);
    const float bv = bias ? bias[ncol] : 0.0f;
#pragma unroll
    for (int r = 0; r < 8; ++r) {
      int m = rowBase + r + hs * 8;
      float v = acc[r] + bv;
      if (ACT == 1) v = (v > 0.0f) ? v : v * slope;
      size_t idx = (size_t)m * Nc + ncol;
      if (ADD) C[idx] += v; else C[idx] = v;
    }
  }
}

// ---------- per-row dot with attention vector: out[r] = X[r,:] . a ----------
__global__ __launch_bounds__(256) void rowdot_kernel(
    const float* __restrict__ X, const float* __restrict__ a,
    float* __restrict__ out, int rows, int cols) {
  const int lane = threadIdx.x & 31;
  const int wave = threadIdx.x >> 5;
  const int r = blockIdx.x * 8 + wave;
  if (r >= rows) return;
  const float* xr = X + (size_t)r * cols;
  float s = 0.0f;
  for (int c = lane; c < cols; c += 32) s += xr[c] * a[c];
#pragma unroll
  for (int off = 16; off > 0; off >>= 1) s += __shfl_xor(s, off, 32);
  if (lane == 0) out[r] = s;
}

// ---------- fills ----------
__global__ void fill_f32(float* p, long n, float v) {
  long i = (long)blockIdx.x * blockDim.x + threadIdx.x;
  if (i < n) p[i] = v;
}
__global__ void fill_u32(unsigned* p, long n, unsigned v) {
  long i = (long)blockIdx.x * blockDim.x + threadIdx.x;
  if (i < n) p[i] = v;
}
__global__ void rowbias_kernel(float* p, const float* __restrict__ bias,
                               long rows, int cols) {
  long i = (long)blockIdx.x * blockDim.x + threadIdx.x;
  if (i < rows * (long)cols) p[i] = bias[i % cols];
}

// ---------- edge pass 1: logits, leaky(0.2), segment max, degrees ----------
__global__ void edge_pass1_kernel(const int* __restrict__ row, const int* __restrict__ col,
                                  const float* __restrict__ an, const float* __restrict__ ae,
                                  float* __restrict__ alpha, unsigned* __restrict__ amax,
                                  float* degn, float* dege, int E, int doDeg) {
  int e = blockIdx.x * blockDim.x + threadIdx.x;
  if (e >= E) return;
  int r = row[e], c = col[e];
  float pre = an[r] + ae[c];
  float al  = (pre > 0.0f) ? pre : 0.2f * pre;
  alpha[e]  = al;
  atomicMax(&amax[c], fenc(al));
  if (doDeg) { atomicAdd(&degn[r], 1.0f); atomicAdd(&dege[c], 1.0f); }
}

// ---------- edge pass 2: ex = exp(alpha - amax[col]); denom[col] += ex ----------
__global__ void edge_pass2_kernel(const int* __restrict__ col, float* __restrict__ alpha,
                                  const unsigned* __restrict__ amax,
                                  float* __restrict__ denom, int E) {
  int e = blockIdx.x * blockDim.x + threadIdx.x;
  if (e >= E) return;
  int c = col[e];
  float m = fdec(amax[c]);
  if (m == -__builtin_inff()) m = 0.0f;  // isfinite guard from reference
  float ex = __expf(alpha[e] - m);
  alpha[e] = ex;
  atomicAdd(&denom[c], ex);
}

// ---------- propagate 1: ef[col] += (ex/denom[col]) * Binv[col] * xl[row] ----------
__global__ __launch_bounds__(256) void prop1_kernel(
    const int* __restrict__ row, const int* __restrict__ col,
    const float* __restrict__ ex, const float* __restrict__ denom,
    const float* __restrict__ dege, const float* __restrict__ X,
    float* __restrict__ ef, int E, int Fd) {
  const int perRow = Fd / 4;
  long w = (long)blockIdx.x * blockDim.x + threadIdx.x;
  if (w >= (long)E * perRow) return;
  int e = (int)(w / perRow);
  int f = (int)(w % perRow) * 4;
  int r = row[e], c = col[e];
  float d    = dege[c];
  float binv = (d > 0.0f) ? 1.0f / d : 0.0f;
  float coef = ex[e] / fmaxf(denom[c], 1e-16f) * binv;
  const float4 xv = *(const float4*)(X + (size_t)r * Fd + f);
  float* o = ef + (size_t)c * Fd + f;
  atomicAdd(o + 0, coef * xv.x);
  atomicAdd(o + 1, coef * xv.y);
  atomicAdd(o + 2, coef * xv.z);
  atomicAdd(o + 3, coef * xv.w);
}

// ---------- propagate 2: out[row] += (ex/denom[col]) * Dinv[row] * ef[col] ----------
__global__ __launch_bounds__(256) void prop2_kernel(
    const int* __restrict__ row, const int* __restrict__ col,
    const float* __restrict__ ex, const float* __restrict__ denom,
    const float* __restrict__ degn, const float* __restrict__ ef,
    float* __restrict__ out, int E, int Fd) {
  const int perRow = Fd / 4;
  long w = (long)blockIdx.x * blockDim.x + threadIdx.x;
  if (w >= (long)E * perRow) return;
  int e = (int)(w / perRow);
  int f = (int)(w % perRow) * 4;
  int r = row[e], c = col[e];
  float d    = degn[r];
  float dinv = (d > 0.0f) ? 1.0f / d : 0.0f;
  float coef = ex[e] / fmaxf(denom[c], 1e-16f) * dinv;
  const float4 xv = *(const float4*)(ef + (size_t)c * Fd + f);
  float* o = out + (size_t)r * Fd + f;
  atomicAdd(o + 0, coef * xv.x);
  atomicAdd(o + 1, coef * xv.y);
  atomicAdd(o + 2, coef * xv.z);
  atomicAdd(o + 3, coef * xv.w);
}

// ---------- GraphNorm per-column stats (block per feature) ----------
__global__ __launch_bounds__(256) void colstats_kernel(
    const float* __restrict__ X, int rows, int cols,
    const float* __restrict__ ms, float* __restrict__ mean, float* __restrict__ rstd) {
  const int f = blockIdx.x;
  float s = 0.0f, ss = 0.0f;
  for (int r = threadIdx.x; r < rows; r += blockDim.x) {
    float v = X[(size_t)r * cols + f];
    s += v; ss += v * v;
  }
  __shared__ float shs[256], shq[256];
  shs[threadIdx.x] = s; shq[threadIdx.x] = ss;
  __syncthreads();
  for (int st = 128; st > 0; st >>= 1) {
    if (threadIdx.x < st) {
      shs[threadIdx.x] += shs[threadIdx.x + st];
      shq[threadIdx.x] += shq[threadIdx.x + st];
    }
    __syncthreads();
  }
  if (threadIdx.x == 0) {
    float m   = shs[0] / rows;
    float ex2 = shq[0] / rows;
    float k   = ms[f];
    // var of (x - k*m) = E[x^2] - 2*k*m*E[x] + k^2*m^2
    float var = ex2 - 2.0f * k * m * m + k * k * m * m;
    mean[f] = m;
    rstd[f] = rsqrtf(var + 1e-5f);
  }
}

// ---------- norm + affine + leaky(0.01) ----------
__global__ void normact_kernel(const float* __restrict__ X,
                               const float* __restrict__ mean, const float* __restrict__ rstd,
                               const float* __restrict__ w, const float* __restrict__ b,
                               const float* __restrict__ ms,
                               float* __restrict__ out, long rows, int cols) {
  long i = (long)blockIdx.x * blockDim.x + threadIdx.x;
  if (i >= rows * (long)cols) return;
  int f = (int)(i % cols);
  float v = (X[i] - ms[f] * mean[f]) * rstd[f] * w[f] + b[f];
  out[i] = (v > 0.0f) ? v : 0.01f * v;
}

// ---------- tiny classifier: logits = O @ clsW + clsb ----------
__global__ void classifier_kernel(const float* __restrict__ O, const float* __restrict__ W,
                                  const float* __restrict__ b, float* __restrict__ out,
                                  int rows, int K, int Oc) {
  int n = blockIdx.x * blockDim.x + threadIdx.x;
  if (n >= rows) return;
  const float* on = O + (size_t)n * K;
  for (int o = 0; o < Oc; ++o) {
    float s = b[o];
    for (int k = 0; k < K; ++k) s += on[k] * W[k * Oc + o];
    out[n * Oc + o] = s;
  }
}

// =====================================================================
static void run_hconv(hipStream_t s,
                      const float* Xin,          // node features [Nn x Ff]
                      const float* ea,           // hyperedge attrs [Mm x Ff]
                      const int* row, const int* col,
                      const _Float16* Wswz, const float* att, const float* bias,
                      float* xl, float* el, float* ef, float* Hout,
                      float* an, float* ae, unsigned* amax, float* denom,
                      float* degn, float* dege, float* alpha, int firstConv) {
  const int TB = 256;
  // linear transforms (WMMA, K = Ff = 512)
  gemm_wmma_kernel<0, 0><<<Nn / 16, TB, 0, s>>>(Xin, Wswz, nullptr, xl, Ff, 0.f);
  gemm_wmma_kernel<0, 0><<<Mm / 16, TB, 0, s>>>(ea,  Wswz, nullptr, el, Ff, 0.f);
  // attention scalars
  rowdot_kernel<<<(Nn + 7) / 8, TB, 0, s>>>(xl, att,      an, Nn, Ff);
  rowdot_kernel<<<(Mm + 7) / 8, TB, 0, s>>>(el, att + Ff, ae, Mm, Ff);
  // init segment-softmax scratch
  fill_u32<<<(Mm + TB - 1) / TB, TB, 0, s>>>(amax, Mm, ENC_NEG_INF);
  fill_f32<<<(Mm + TB - 1) / TB, TB, 0, s>>>(denom, Mm, 0.0f);
  if (firstConv) {
    fill_f32<<<(Nn + TB - 1) / TB, TB, 0, s>>>(degn, Nn, 0.0f);
    fill_f32<<<(Mm + TB - 1) / TB, TB, 0, s>>>(dege, Mm, 0.0f);
  }
  edge_pass1_kernel<<<(Ee + TB - 1) / TB, TB, 0, s>>>(row, col, an, ae, alpha, amax,
                                                      degn, dege, Ee, firstConv);
  edge_pass2_kernel<<<(Ee + TB - 1) / TB, TB, 0, s>>>(col, alpha, amax, denom, Ee);
  // propagate 1: nodes -> hyperedges
  fill_f32<<<((long)Mm * Ff + TB - 1) / TB, TB, 0, s>>>(ef, (long)Mm * Ff, 0.0f);
  {
    long tot = (long)Ee * (Ff / 4);
    prop1_kernel<<<(unsigned)((tot + TB - 1) / TB), TB, 0, s>>>(row, col, alpha, denom,
                                                                dege, xl, ef, Ee, Ff);
  }
  // propagate 2: hyperedges -> nodes (Hout pre-seeded with bias)
  rowbias_kernel<<<((long)Nn * Ff + TB - 1) / TB, TB, 0, s>>>(Hout, bias, Nn, Ff);
  {
    long tot = (long)Ee * (Ff / 4);
    prop2_kernel<<<(unsigned)((tot + TB - 1) / TB), TB, 0, s>>>(row, col, alpha, denom,
                                                                degn, ef, Hout, Ee, Ff);
  }
}

extern "C" void kernel_launch(void* const* d_in, const int* in_sizes, int n_in,
                              void* d_out, int out_size, void* d_ws, size_t ws_size,
                              hipStream_t stream) {
  const float* x    = (const float*)d_in[0];
  const int*   ei   = (const int*)d_in[1];
  const float* ea   = (const float*)d_in[2];
  const float* W1   = (const float*)d_in[3];
  const float* att1 = (const float*)d_in[4];
  const float* b1   = (const float*)d_in[5];
  const float* W2   = (const float*)d_in[6];
  const float* att2 = (const float*)d_in[7];
  const float* b2   = (const float*)d_in[8];
  const float* gn1w = (const float*)d_in[9];
  const float* gn1b = (const float*)d_in[10];
  const float* gn1ms= (const float*)d_in[11];
  const float* gn2w = (const float*)d_in[12];
  const float* gn2b = (const float*)d_in[13];
  const float* gn2ms= (const float*)d_in[14];
  const float* fc1W = (const float*)d_in[15];
  const float* fc1b = (const float*)d_in[16];
  const float* fc2W = (const float*)d_in[17];
  const float* fc2b = (const float*)d_in[18];
  const float* clsW = (const float*)d_in[19];
  const float* clsb = (const float*)d_in[20];
  const int* row = ei;
  const int* col = ei + Ee;
  float* out = (float*)d_out;

  // workspace carve-up (all block sizes are multiples of 8 floats -> 32B aligned)
  float* ws = (float*)d_ws;
  size_t off = 0;
  auto alloc = [&](size_t n) { float* p = ws + off; off += n; return p; };
  float* bufA  = alloc((size_t)Nn * Ff);   // xl1 -> G (normed h1) -> conv2 output
  float* bufEl = alloc((size_t)Mm * Ff);   // el1 -> el2
  float* bufEf = alloc((size_t)Mm * Ff);   // edge_feat scatter target
  float* bufH  = alloc((size_t)Nn * Ff);   // conv1 out -> xl2 -> normed h2
  float* bufO  = alloc((size_t)Nn * HIDd); // fc accumulator
  float* alpha = alloc(Ee);
  float* an    = alloc(Nn);
  float* ae    = alloc(Mm);
  unsigned* amax = (unsigned*)alloc(Mm);
  float* denom = alloc(Mm);
  float* degn  = alloc(Nn);
  float* dege  = alloc(Mm);
  float* meanb = alloc(Ff);
  float* rstdb = alloc(Ff);
  _Float16* W1s  = (_Float16*)alloc((size_t)Ff * Ff / 2);
  _Float16* W2s  = (_Float16*)alloc((size_t)Ff * Ff / 2);
  _Float16* fc1s = (_Float16*)alloc((size_t)Ff * HIDd / 2);
  _Float16* fc2s = (_Float16*)alloc((size_t)Ff * HIDd / 2);
  (void)ws_size; (void)in_sizes; (void)n_in; (void)out_size;

  const int TB = 256;

  // pre-swizzle weights to f16 WMMA B-fragments
  {
    int tW  = (Ff / 16) * (Ff / 32) * 32;    // 16384
    int tFc = (HIDd / 16) * (Ff / 32) * 32;  // 8192
    prep_b_kernel<<<(tW + TB - 1) / TB, TB, 0, stream>>>(W1,   W1s,  Ff, Ff);
    prep_b_kernel<<<(tW + TB - 1) / TB, TB, 0, stream>>>(W2,   W2s,  Ff, Ff);
    prep_b_kernel<<<(tFc + TB - 1) / TB, TB, 0, stream>>>(fc1W, fc1s, Ff, HIDd);
    prep_b_kernel<<<(tFc + TB - 1) / TB, TB, 0, stream>>>(fc2W, fc2s, Ff, HIDd);
  }

  // ===== layer 1: hypergraph conv =====
  run_hconv(stream, x, ea, row, col, W1s, att1, b1,
            bufA, bufEl, bufEf, bufH, an, ae, amax, denom, degn, dege, alpha, 1);
  // GraphNorm1 + leaky -> G (reuse bufA; xl1 no longer needed)
  colstats_kernel<<<Ff, TB, 0, stream>>>(bufH, Nn, Ff, gn1ms, meanb, rstdb);
  normact_kernel<<<((long)Nn * Ff + TB - 1) / TB, TB, 0, stream>>>(
      bufH, meanb, rstdb, gn1w, gn1b, gn1ms, bufA, Nn, Ff);
  // fc1: O = leaky(G @ fc1W + fc1b)
  gemm_wmma_kernel<1, 0><<<Nn / 16, TB, 0, stream>>>(bufA, fc1s, fc1b, bufO, HIDd, 0.01f);

  // ===== layer 2: hypergraph conv on G =====
  run_hconv(stream, bufA, ea, row, col, W2s, att2, b2,
            bufH /*xl2*/, bufEl, bufEf, bufA /*conv2 out*/,
            an, ae, amax, denom, degn, dege, alpha, 0);
  // GraphNorm2 + leaky -> bufH (xl2 free after propagation)
  colstats_kernel<<<Ff, TB, 0, stream>>>(bufA, Nn, Ff, gn2ms, meanb, rstdb);
  normact_kernel<<<((long)Nn * Ff + TB - 1) / TB, TB, 0, stream>>>(
      bufA, meanb, rstdb, gn2w, gn2b, gn2ms, bufH, Nn, Ff);
  // fc2: O += leaky(h2n @ fc2W + fc2b)
  gemm_wmma_kernel<1, 1><<<Nn / 16, TB, 0, stream>>>(bufH, fc2s, fc2b, bufO, HIDd, 0.01f);

  // ===== classifier =====
  classifier_kernel<<<(Nn + TB - 1) / TB, TB, 0, stream>>>(bufO, clsW, clsb, out,
                                                           Nn, HIDd, OUTc);
}
